// CommNet_61675730371151
// MI455X (gfx1250) — matrix operand order repaired and checked by reading, not compiled, not run on hardware
//
#include <hip/hip_runtime.h>
#include <hip/hip_fp16.h>

// ---------------------------------------------------------------------------
// CommNet fused forward for gfx1250 (MI455X), wave32 + V_WMMA_F32_16X16X32_F16
// ---------------------------------------------------------------------------
typedef __attribute__((ext_vector_type(16))) _Float16 v16h;
typedef __attribute__((ext_vector_type(8)))  float    v8f;

union Frag16 { v16h h; uint4 q[2]; };
union H8     { uint4 q; _Float16 v[8]; };
union H4     { uint2 q; _Float16 v[4]; };

constexpr int O_ = 128;     // obs feature dim
constexpr int H_ = 256;     // hidden dim
constexpr int A_ = 8;       // action dim
constexpr int MB = 128;     // rows per workgroup (4 batches x 32 agents)
constexpr int HS = 264;     // h LDS row stride (halves), padded
constexpr int OS = 136;     // obs LDS row stride (halves), padded
constexpr int SS = 264;     // S LDS row stride (halves)
constexpr long long BNROWS = 4096LL * 32LL;   // 131072 total rows

// Branch-free tanh: 1 - 2/(1+2^(2*log2(e)*x)).  v_exp_f32 + v_rcp_f32, no
// divergence; saturates to +/-1 via exp2 -> {inf,0} for large |x|.
__device__ __forceinline__ float fast_tanh(float x) {
    float e = __builtin_amdgcn_exp2f(x * 2.8853900817779268f);
    return 1.0f - 2.0f * __builtin_amdgcn_rcpf(e + 1.0f);
}

// fragment-packed offset for a (k, n) weight element.
// B-fragment layout (16x16x32 f16): lane = (n%16) + 16*((k%32)/16),
// half index t = k%16; lane-contiguous 16 halves (=32B) per fragment.
__device__ __host__ inline int pack_off(int k, int n) {
    int kt = k >> 5, kl = k & 31;
    int nt = n >> 4, nl = n & 15;
    int lane = nl + ((kl >> 4) << 4);
    int t    = kl & 15;
    return ((kt * 16 + nt) << 9) + (lane << 4) + t;
}

// ---------------------------------------------------------------------------
// Prep kernel: f32 -> f16 weight conversion, comm-folded weights, WMMA packing
//   pWf  = pack(W_f)            (iteration 0: c == 0)
//   pW2  = pack(W_f - W_C/31)   (iterations 1,2: self-exclusion folded in)
//   pWc  = pack(W_C/31)         (for t = S @ W_C/31 rank correction)
//   pWe  = pack(W_enc)
// ---------------------------------------------------------------------------
__global__ void prep_weights(const float* __restrict__ W_enc,
                             const float* __restrict__ W_f,
                             const float* __restrict__ W_C,
                             _Float16* __restrict__ pWe,
                             _Float16* __restrict__ pWf,
                             _Float16* __restrict__ pW2,
                             _Float16* __restrict__ pWc) {
    int gid = blockIdx.x * blockDim.x + threadIdx.x;
    if (gid >= H_ * H_) return;
    int k = gid >> 8;
    int n = gid & 255;
    float wf  = W_f[gid];
    float wc  = W_C[gid];
    float c31 = wc * (1.0f / 31.0f);
    int off = pack_off(k, n);
    pWf[off] = (_Float16)wf;
    pW2[off] = (_Float16)(wf - c31);
    pWc[off] = (_Float16)c31;
    if (k < O_) pWe[off] = (_Float16)W_enc[k * H_ + n];
}

__device__ __forceinline__ void load_bfrag(Frag16& b, const _Float16* Bp,
                                           int kc, int nt, int lane) {
    const _Float16* bp = Bp + ((kc * 16 + nt) << 9) + (lane << 4);
    b.q[0] = *(const uint4*)(bp);
    b.q[1] = *(const uint4*)(bp + 8);
}

// ---------------------------------------------------------------------------
// Per-wave 16-row GEMM: D(16x256) = tanh(A(16xK) @ B(Kx256) + bias [+ t])
// A fragments are fully hoisted to registers first => in-place LDS update OK.
// 4 column tiles per iteration, each with double-buffered B fragments, so the
// scheduler keeps 4 independent load streams in flight over the WMMAs.
// Accumulators start from inline-zero C; bias (+ comm term) is folded into the
// epilogue tanh input, keeping the WMMA critical path free of broadcasts.
// ---------------------------------------------------------------------------
template <int KT, bool HAS_T>
__device__ __forceinline__ void wave_gemm_tanh(const _Float16* As, int strideA,
                                               const _Float16* __restrict__ Bp,
                                               const float* biasCol,
                                               const float* tRow,
                                               _Float16* Dout, int lane) {
    const int hs = lane >> 4;   // half-wave select
    const int lm = lane & 15;

    Frag16 a[KT];
#pragma unroll
    for (int kc = 0; kc < KT; ++kc) {
        const _Float16* ar = As + lm * strideA + kc * 32 + hs * 16;
        a[kc].q[0] = *(const uint4*)(ar);        // K = base .. base+7
        a[kc].q[1] = *(const uint4*)(ar + 8);    // K = base+8 .. base+15
    }

#pragma unroll 1
    for (int np = 0; np < 4; ++np) {             // 4 column tiles / iteration
        const int ntb = np * 4;
        if (np + 1 < 4)                           // warm caches for next group
            __builtin_prefetch(Bp + ((ntb + 4) << 9), 0, 1);

        float init[4];
#pragma unroll
        for (int u = 0; u < 4; ++u) {
            const int col = ((ntb + u) << 4) + lm;
            init[u] = biasCol[col];
            if (HAS_T) init[u] += tRow[col];
        }

        Frag16 b[4][2];                           // 4 streams, double-buffered
#pragma unroll
        for (int u = 0; u < 4; ++u) load_bfrag(b[u][0], Bp, 0, ntb + u, lane);

        v8f acc[4];
#pragma unroll
        for (int u = 0; u < 4; ++u) acc[u] = (v8f){};   // inline-zero C

#pragma unroll
        for (int kc = 0; kc < KT; ++kc) {
            const int cur = kc & 1, nxt = cur ^ 1;
            if (kc + 1 < KT) {
#pragma unroll
                for (int u = 0; u < 4; ++u)
                    load_bfrag(b[u][nxt], Bp, kc + 1, ntb + u, lane);
            }
#pragma unroll
            for (int u = 0; u < 4; ++u)
                acc[u] = __builtin_amdgcn_wmma_f32_16x16x32_f16(
                    false, a[kc].h, false, b[u][cur].h, (short)0, acc[u],
                    false, false);
        }

#pragma unroll
        for (int r = 0; r < 8; ++r) {
            const int row = r + (hs << 3);
#pragma unroll
            for (int u = 0; u < 4; ++u) {
                const int col = ((ntb + u) << 4) + lm;
                Dout[row * HS + col] =
                    (_Float16)fast_tanh(acc[u][r] + init[u]);
            }
        }
    }
}

// ---------------------------------------------------------------------------
// Fully fused CommNet forward: encoder -> 3 comm iterations -> heads.
// One workgroup = 128 rows (4 batches). h never leaves LDS.
// ---------------------------------------------------------------------------
__global__ void __launch_bounds__(256)
commnet_fused(const float* __restrict__ obs,
              const float* __restrict__ b_enc,
              const float* __restrict__ b_f,
              const float* __restrict__ b_C,
              const float* __restrict__ W_mean,
              const float* __restrict__ b_mean,
              const float* __restrict__ log_std,
              const float* __restrict__ W_v,
              const float* __restrict__ b_v,
              const _Float16* __restrict__ pWe,
              const _Float16* __restrict__ pWf,
              const _Float16* __restrict__ pW2,
              const _Float16* __restrict__ pWc,
              float* __restrict__ out) {
    extern __shared__ char smem[];
    _Float16* hL   = (_Float16*)smem;          // 128 x 264 halves
    _Float16* obsL = hL + MB * HS;             // 128 x 136 halves
    _Float16* sL   = obsL + MB * OS;           // 16 x 264 halves (rows 4..15 = 0)
    float*    tL   = (float*)(sL + 16 * SS);   // 4 x 256 f32
    float*    wmL  = tL + 4 * H_;              // 256 x 8 f32
    float*    wvL  = wmL + H_ * A_;            // 256 f32
    float*    bencL  = wvL + H_;               // 256 f32
    float*    biterL = bencL + H_;             // 256 f32 (b_f + b_C)
    float*    lsL  = biterL + H_;              // 8 f32

    const int tid  = threadIdx.x;
    const int wid  = tid >> 5;                 // wave 0..7 (wave32)
    const int lane = tid & 31;
    const int row0 = blockIdx.x * MB;

    // ---------------- phase 0: stage obs (f32 -> f16) + small params --------
    {
        int row = tid >> 1, hp = tid & 1;
        const float* src = obs + (size_t)(row0 + row) * O_ + hp * 64;
        _Float16*    dst = obsL + row * OS + hp * 64;
#pragma unroll
        for (int i = 0; i < 16; ++i) {
            float4 v = *(const float4*)(src + 4 * i);
            dst[4 * i + 0] = (_Float16)v.x;
            dst[4 * i + 1] = (_Float16)v.y;
            dst[4 * i + 2] = (_Float16)v.z;
            dst[4 * i + 3] = (_Float16)v.w;
        }
        for (int i = tid; i < H_ * A_; i += 256) wmL[i] = W_mean[i];
        if (tid < H_) {
            wvL[tid]    = W_v[tid];
            bencL[tid]  = b_enc[tid];
            biterL[tid] = b_f[tid] + b_C[tid];
        }
        if (tid < A_) lsL[tid] = log_std[tid];
        for (int i = tid; i < 16 * SS; i += 256) sL[i] = (_Float16)0.0f;
    }
    __syncthreads();

    // ---------------- encoder: h = tanh(obs @ W_enc + b_enc) ----------------
    wave_gemm_tanh<4, false>(obsL + wid * 16 * OS, OS, pWe, bencL, nullptr,
                             hL + wid * 16 * HS, lane);
    __syncthreads();

    // ---------------- iteration 0 (c == 0): h = tanh(h@W_f + b_f + b_C) -----
    wave_gemm_tanh<8, false>(hL + wid * 16 * HS, HS, pWf, biterL, nullptr,
                             hL + wid * 16 * HS, lane);
    __syncthreads();

    // ---------------- iterations 1,2 ----------------------------------------
    for (int it = 1; it < 3; ++it) {
        // per-batch column sums S[b][:] = sum_agents h[b,agent,:]
        {
            int b  = tid >> 6;       // batch 0..3
            int c0 = (tid & 63) * 4; // 4-column group
            float s0 = 0, s1 = 0, s2 = 0, s3 = 0;
            const _Float16* hb = hL + (b * 32) * HS + c0;
            for (int a2 = 0; a2 < 32; ++a2) {
                H4 v; v.q = *(const uint2*)(hb + a2 * HS);
                s0 += (float)v.v[0]; s1 += (float)v.v[1];
                s2 += (float)v.v[2]; s3 += (float)v.v[3];
            }
            _Float16* sp = sL + b * SS + c0;
            sp[0] = (_Float16)s0; sp[1] = (_Float16)s1;
            sp[2] = (_Float16)s2; sp[3] = (_Float16)s3;
        }
        __syncthreads();
        // t = S @ (W_C/31): one 16-row WMMA GEMM, 2 col-tiles per wave,
        // double-buffered like the main GEMM.
        {
            const int hs = lane >> 4, lm = lane & 15;
            const int nt0 = wid * 2, nt1 = nt0 + 1;
            v8f acc0 = {}, acc1 = {};
            Frag16 b0[2], b1[2];
            load_bfrag(b0[0], pWc, 0, nt0, lane);
            load_bfrag(b1[0], pWc, 0, nt1, lane);
#pragma unroll
            for (int kc = 0; kc < 8; ++kc) {
                Frag16 afr;
                const _Float16* ar = sL + lm * SS + kc * 32 + hs * 16;
                afr.q[0] = *(const uint4*)(ar);
                afr.q[1] = *(const uint4*)(ar + 8);
                const int cur = kc & 1, nxt = cur ^ 1;
                if (kc + 1 < 8) {
                    load_bfrag(b0[nxt], pWc, kc + 1, nt0, lane);
                    load_bfrag(b1[nxt], pWc, kc + 1, nt1, lane);
                }
                acc0 = __builtin_amdgcn_wmma_f32_16x16x32_f16(
                    false, afr.h, false, b0[cur].h, (short)0, acc0, false, false);
                acc1 = __builtin_amdgcn_wmma_f32_16x16x32_f16(
                    false, afr.h, false, b1[cur].h, (short)0, acc1, false, false);
            }
            if (hs == 0) {   // rows 0..3 = batches 0..3
#pragma unroll
                for (int r = 0; r < 4; ++r) {
                    tL[r * H_ + nt0 * 16 + lm] = acc0[r];
                    tL[r * H_ + nt1 * 16 + lm] = acc1[r];
                }
            }
        }
        __syncthreads();
        // h = tanh(h @ (W_f - W_C/31) + (b_f + b_C) + t)  (in place)
        wave_gemm_tanh<8, true>(hL + wid * 16 * HS, HS, pW2, biterL,
                                tL + (wid >> 1) * H_, hL + wid * 16 * HS, lane);
        __syncthreads();
    }

    // ---------------- heads (tiny): mean/value + broadcast std --------------
    if (tid < MB) {
        int row   = tid;
        size_t rg = (size_t)row0 + row;
        float am[A_];
#pragma unroll
        for (int a = 0; a < A_; ++a) am[a] = b_mean[a];
        float av = b_v[0];
        const _Float16* hr = hL + row * HS;
        for (int kc = 0; kc < 32; ++kc) {
            H8 v; v.q = *(const uint4*)(hr + kc * 8);
#pragma unroll
            for (int j = 0; j < 8; ++j) {
                float hv = (float)v.v[j];
                int   k  = kc * 8 + j;
#pragma unroll
                for (int a = 0; a < A_; ++a) am[a] += hv * wmL[k * A_ + a];
                av += hv * wvL[k];
            }
        }
        float* om   = out;
        float* ols  = out + (size_t)BNROWS * A_;
        float* ostd = ols + (size_t)BNROWS * A_;
        float* oval = ostd + (size_t)BNROWS * A_;
#pragma unroll
        for (int a = 0; a < A_; ++a) {
            om[rg * A_ + a] = am[a];
            float lsv       = lsL[a];
            ols[rg * A_ + a]  = lsv;
            ostd[rg * A_ + a] = expf(lsv);
        }
        oval[rg] = av;
    }
}

// ---------------------------------------------------------------------------
extern "C" void kernel_launch(void* const* d_in, const int* in_sizes, int n_in,
                              void* d_out, int out_size, void* d_ws,
                              size_t ws_size, hipStream_t stream) {
    (void)in_sizes; (void)n_in; (void)out_size; (void)ws_size;

    const float* obs     = (const float*)d_in[0];
    const float* W_enc   = (const float*)d_in[1];
    const float* b_enc   = (const float*)d_in[2];
    const float* W_f     = (const float*)d_in[3];
    const float* b_f     = (const float*)d_in[4];
    const float* W_C     = (const float*)d_in[5];
    const float* b_C     = (const float*)d_in[6];
    const float* W_mean  = (const float*)d_in[7];
    const float* b_mean  = (const float*)d_in[8];
    const float* log_std = (const float*)d_in[9];
    const float* W_v     = (const float*)d_in[10];
    const float* b_v     = (const float*)d_in[11];

    _Float16* pWe = (_Float16*)d_ws;          // 128*256 halves
    _Float16* pWf = pWe + 128 * 256;          // 256*256 halves
    _Float16* pW2 = pWf + 256 * 256;          // 256*256 halves
    _Float16* pWc = pW2 + 256 * 256;          // 256*256 halves (total ~448 KB)

    prep_weights<<<256, 256, 0, stream>>>(W_enc, W_f, W_C, pWe, pWf, pW2, pWc);

    constexpr size_t kHalfBytes = (size_t)(MB * HS + MB * OS + 16 * SS) * 2;
    constexpr size_t kFloatBytes =
        (size_t)(4 * H_ + H_ * A_ + H_ + H_ + H_ + A_) * 4;
    constexpr size_t kSmemBytes = kHalfBytes + kFloatBytes;  // ~124 KB

    commnet_fused<<<(int)(BNROWS / MB), 256, kSmemBytes, stream>>>(
        obs, b_enc, b_f, b_C, W_mean, b_mean, log_std, W_v, b_v,
        pWe, pWf, pW2, pWc, (float*)d_out);
}
